// AFFWithCustomGCN_35072702939294
// MI455X (gfx1250) — matrix operand order repaired and checked by reading, not compiled, not run on hardware
//
#include <hip/hip_runtime.h>
#include <hip/hip_bf16.h>

typedef __attribute__((ext_vector_type(16))) _Float16 v16h;
typedef __attribute__((ext_vector_type(8)))  _Float16 v8h;
typedef __attribute__((ext_vector_type(8)))  float    v8f;

#define NB    256   // batch
#define CC    32    // channels (in == out)
#define TTOT  128   // frames
#define VV    25    // joints
#define HH    3     // heads
#define VP    32    // padded joints
#define TTILE 16    // frames per block
#define ROWS  (CC*TTILE)   // 512 = GEMM1 M

__global__ __launch_bounds__(256) void sagc_fused_kernel(
    const float* __restrict__ x,     const float* __restrict__ A,
    const float* __restrict__ attn,  const float* __restrict__ Wd,
    const float* __restrict__ bd,    const float* __restrict__ bn_g,
    const float* __restrict__ bn_b,  const float* __restrict__ bn_m,
    const float* __restrict__ bn_v,  float* __restrict__ out)
{
    __shared__ alignas(64) _Float16 Xf[ROWS * VP];      // 32KB  [row=c*16+t][v]
    __shared__ alignas(64) _Float16 AhT[HH * VP * VP];  // 6KB   [h][w][v]  (transposed A*attn)
    __shared__ alignas(64) _Float16 Wdl[HH * CC * CC];  // 6KB   [h][o][c]
    __shared__ alignas(64) _Float16 featT[ROWS * CC];   // 32KB  [p=t*32+w][c]
    __shared__ float bsum[CC], binv[CC], bbeta[CC], bmean[CC];

    const int tid  = threadIdx.x;
    const int lane = tid & 31;
    const int wv   = tid >> 5;       // wave 0..7
    const int l16  = lane & 15;
    const int half = lane >> 4;

    const int n  = blockIdx.x >> 3;          // batch index
    const int t0 = (blockIdx.x & 7) * TTILE; // frame tile

    // ---- stage x tile (fp32 -> f16, pad V to 32) ----
    for (int e = tid; e < ROWS * VP; e += 256) {
        int v = e & 31, row = e >> 5;
        int c = row >> 4, t = row & 15;
        float val = (v < VV) ? x[((n * CC + c) * TTOT + t0 + t) * VV + v] : 0.f;
        Xf[e] = (_Float16)val;
    }
    // ---- AhT[h][w][v] = A[h][v][w] * attn[n][h][v][w], zero padded ----
    for (int e = tid; e < HH * VP * VP; e += 256) {
        int v = e & 31, w = (e >> 5) & 31, h = e >> 10;
        float val = 0.f;
        if (v < VV && w < VV) {
            int i2 = (h * VV + v) * VV + w;
            val = A[i2] * attn[n * (HH * VV * VV) + i2];
        }
        AhT[e] = (_Float16)val;
    }
    // ---- Wd [h][o][c] fp32 -> f16 ----
    for (int e = tid; e < HH * CC * CC; e += 256)
        Wdl[e] = (_Float16)Wd[e];
    // ---- per-channel epilogue constants ----
    if (tid < CC) {
        bsum[tid]  = bd[tid] + bd[CC + tid] + bd[2 * CC + tid];
        binv[tid]  = bn_g[tid] * rsqrtf(bn_v[tid] + 1e-5f);
        bbeta[tid] = bn_b[tid];
        bmean[tid] = bn_m[tid];
    }

    v8f acc[8];
    #pragma unroll
    for (int i = 0; i < 8; ++i) acc[i] = (v8f){};

    const int mt2 = wv & 1;          // GEMM2 o-tile
    const int ntb = (wv >> 1) * 8;   // GEMM2 p-tile base

    for (int h = 0; h < HH; ++h) {
        __syncthreads();   // LDS staged / featT free from previous head

        // ---- GEMM1: feat[(c,t), w] = sum_v Xf[(c,t), v] * Ah[h][v][w]
        //      per wave: 4 distinct A-frags x 2 distinct B-frags -> 8 WMMAs.
        //      Hoist the 6 fragment loads so each LDS line is read exactly once.
        v16h a1[4];
        #pragma unroll
        for (int j = 0; j < 4; ++j) {
            int row = (((wv << 2) + j) << 4) + l16;
            v8h lo = *(const v8h*)&Xf[row * VP + half * 16];
            v8h hi = *(const v8h*)&Xf[row * VP + half * 16 + 8];
            a1[j] = __builtin_shufflevector(lo, hi,
                0,1,2,3,4,5,6,7,8,9,10,11,12,13,14,15);
        }
        v16h b1[2];
        #pragma unroll
        for (int j = 0; j < 2; ++j) {
            int wcol = (j << 4) + l16;
            b1[j] = *(const v16h*)&AhT[(h * VP + wcol) * VP + half * 16];
        }
        #pragma unroll
        for (int i = 0; i < 8; ++i) {
            int mt = (wv << 2) + (i & 3);   // 0..31 (== channel c)
            int nt = i >> 2;                // 0..1
            v8f d = (v8f){};
            d = __builtin_amdgcn_wmma_f32_16x16x32_f16(
                    false, a1[i & 3], false, b1[nt], (short)0, d, false, false);
            // scatter D -> featT[(t,w)][c], c == mt
            int wcol = (nt << 4) + l16;
            #pragma unroll
            for (int r = 0; r < 8; ++r) {
                int tl = r + (half << 3);
                featT[(tl * VP + wcol) * CC + mt] = (_Float16)d[r];
            }
        }
        __syncthreads();   // featT ready

        // ---- GEMM2: out[o, p] += sum_c Wd[h][o][c] * feat[c][p]  (K = 32)
        {
            int orow = (h * CC + (mt2 << 4) + l16) * CC;
            v8h wlo = *(const v8h*)&Wdl[orow + half * 16];
            v8h whi = *(const v8h*)&Wdl[orow + half * 16 + 8];
            v16h wfrag = __builtin_shufflevector(wlo, whi,
                0,1,2,3,4,5,6,7,8,9,10,11,12,13,14,15);
            #pragma unroll
            for (int i = 0; i < 8; ++i) {
                int p = ((ntb + i) << 4) + l16;
                v16h bfrag = *(const v16h*)&featT[p * CC + half * 16];
                acc[i] = __builtin_amdgcn_wmma_f32_16x16x32_f16(
                            false, wfrag, false, bfrag, (short)0, acc[i], false, false);
            }
        }
    }

    // ---- epilogue: +bias, BN(inference), +x residual, ReLU (relu∘relu == relu) ----
    #pragma unroll
    for (int i = 0; i < 8; ++i) {
        int nt   = ntb + i;
        int tl   = nt >> 1;
        int wcol = ((nt & 1) << 4) + l16;
        if (wcol < VV) {
            #pragma unroll
            for (int r = 0; r < 8; ++r) {
                int o = (mt2 << 4) + r + (half << 3);
                float val = acc[i][r] + bsum[o];
                val = (val - bmean[o]) * binv[o] + bbeta[o];
                int gi = ((n * CC + o) * TTOT + t0 + tl) * VV + wcol;
                val += x[gi];                 // identity residual (L2-resident re-read)
                out[gi] = fmaxf(val, 0.f);
            }
        }
    }
}

extern "C" void kernel_launch(void* const* d_in, const int* in_sizes, int n_in,
                              void* d_out, int out_size, void* d_ws, size_t ws_size,
                              hipStream_t stream) {
    const float* x    = (const float*)d_in[0];
    const float* A    = (const float*)d_in[1];
    const float* attn = (const float*)d_in[2];
    const float* Wd   = (const float*)d_in[3];
    const float* bd   = (const float*)d_in[4];
    const float* bn_g = (const float*)d_in[5];
    const float* bn_b = (const float*)d_in[6];
    const float* bn_m = (const float*)d_in[7];
    const float* bn_v = (const float*)d_in[8];
    float* out = (float*)d_out;

    dim3 grid(NB * (TTOT / TTILE));   // 2048 blocks: (n, t-tile)
    dim3 block(256);                  // 8 wave32
    sagc_fused_kernel<<<grid, block, 0, stream>>>(
        x, A, attn, Wd, bd, bn_g, bn_b, bn_m, bn_v, out);
}
